// MoEKANConvBase_70866960384442
// MI455X (gfx1250) — compile-verified, matrix operand order checked by reading, not run on hardware
//
#include <hip/hip_runtime.h>

// ---------------- types ----------------
typedef __attribute__((ext_vector_type(16))) __bf16 v16bf;
typedef __attribute__((ext_vector_type(8), aligned(16))) __bf16 v8bf;
typedef __attribute__((ext_vector_type(8))) float v8f;

static __device__ inline unsigned short f2bf(float f) {
  union { float f; unsigned u; } v; v.f = f;
  unsigned r = v.u + 0x7FFFu + ((v.u >> 16) & 1u);   // round-to-nearest-even
  return (unsigned short)(r >> 16);
}

// Build a v16bf WMMA fragment from two contiguous 16-byte groups:
// halves 0..7  <- p[0..7]   (k = 8*hi + 0..7)
// halves 8..15 <- p[16..23] (k = 8*hi + 16..23)
static __device__ inline v16bf mk16(const unsigned short* p) {
  union { v16bf v; struct { v8bf lo; v8bf hi; } s; } u;
  u.s.lo = *(const v8bf*)(p);
  u.s.hi = *(const v8bf*)(p + 16);
  return u.v;
}

// ---------------- constants ----------------
#define BATCH 16
#define CIN   64
#define COUT  64
#define HH    64
#define WW    64
#define NEXP  16
#define KTOP  4
// LDS slab: 4 input rows x 66 (w, halo) x 72 (ci padded from 64) bf16 = 38016 B
#define CI_STRIDE 72
#define XS_ELEMS (4 * 66 * CI_STRIDE)

// ---------------- K1: gate_x = mean over H,W ----------------
__global__ __launch_bounds__(256) void k_gatex(const float* __restrict__ x,
                                               float* __restrict__ gate_x) {
  __shared__ float red[256];
  const int tid = threadIdx.x;
  const long base = (long)blockIdx.x * (HH * WW);
  float s = 0.f;
  for (int i = tid; i < HH * WW; i += 256) s += x[base + i];
  red[tid] = s;
  __syncthreads();
  for (int off = 128; off > 0; off >>= 1) {
    if (tid < off) red[tid] += red[tid + off];
    __syncthreads();
  }
  if (tid == 0) gate_x[blockIdx.x] = red[0] * (1.0f / (HH * WW));
}

// ---------------- K2: gating (logits, softmax, top-4, loss) ----------------
__global__ __launch_bounds__(256) void k_gate(const float* __restrict__ gate_x,
                                              const float* __restrict__ w_gate,
                                              float* __restrict__ gates_out,
                                              float* __restrict__ loss_out) {
  __shared__ float cl[BATCH][NEXP];
  __shared__ float gsh[BATCH][NEXP];
  const int t = threadIdx.x;
  const int bb = t >> 4, e = t & 15;
  gsh[bb][e] = 0.f;
  float s = 0.f;
  for (int ci = 0; ci < CIN; ++ci) s += gate_x[bb * CIN + ci] * w_gate[ci * NEXP + e];
  cl[bb][e] = s;
  __syncthreads();

  if (t < BATCH) {
    float p[NEXP];
    float m = -1e30f;
    for (int i = 0; i < NEXP; ++i) m = fmaxf(m, cl[t][i]);
    float den = 0.f;
    for (int i = 0; i < NEXP; ++i) { p[i] = expf(cl[t][i] - m); den += p[i]; }
    float inv = 1.f / den;
    for (int i = 0; i < NEXP; ++i) p[i] *= inv;
    // top-4 (first-index tie break, like jax top_k)
    float rem[NEXP];
    for (int i = 0; i < NEXP; ++i) rem[i] = p[i];
    int idx4[KTOP]; float val4[KTOP]; float sum4 = 0.f;
    for (int k = 0; k < KTOP; ++k) {
      int best = 0; float bv = rem[0];
      for (int i = 1; i < NEXP; ++i) if (rem[i] > bv) { bv = rem[i]; best = i; }
      idx4[k] = best; val4[k] = bv; rem[best] = -1e30f; sum4 += bv;
    }
    float invs = 1.f / (sum4 + 1e-6f);
    for (int k = 0; k < KTOP; ++k) gsh[t][idx4[k]] = val4[k] * invs;
  }
  __syncthreads();
  gates_out[t] = gsh[bb][e];
  if (t == 0) {
    float imp[NEXP], ld[NEXP];
    for (int i = 0; i < NEXP; ++i) {
      float si = 0.f, sl = 0.f;
      for (int b2 = 0; b2 < BATCH; ++b2) {
        si += gsh[b2][i];
        sl += (gsh[b2][i] > 0.f) ? 1.f : 0.f;
      }
      imp[i] = si; ld[i] = sl;
    }
    float loss = 0.f;
    for (int pass = 0; pass < 2; ++pass) {
      const float* v = pass ? ld : imp;
      float mean = 0.f;
      for (int i = 0; i < NEXP; ++i) mean += v[i];
      mean *= (1.f / NEXP);
      float var = 0.f;
      for (int i = 0; i < NEXP; ++i) { float d = v[i] - mean; var += d * d; }
      var *= (1.f / (NEXP - 1));       // ddof=1
      loss += var / (mean * mean + 1e-10f);
    }
    *loss_out = loss * 0.01f;
  }
}

// ---------------- K3: effective bias ----------------
__global__ __launch_bounds__(256) void k_beff(const float* __restrict__ bias,
                                              const float* __restrict__ gates,
                                              float* __restrict__ beff) {
  const int idx = blockIdx.x * 256 + threadIdx.x;     // 16*64
  const int b = idx >> 6, co = idx & 63;
  float s = 0.f;
  for (int e = 0; e < NEXP; ++e) s += gates[b * NEXP + e] * bias[e * COUT + co];
  beff[idx] = s;
}

// ---------------- K4: effective weights, bf16, layout [b][tap][co][ci] ----------------
__global__ __launch_bounds__(256) void k_weff(const float* __restrict__ W,
                                              const float* __restrict__ gates,
                                              unsigned short* __restrict__ weff) {
  __shared__ float g[NEXP];
  const int idx = blockIdx.x * 256 + threadIdx.x;     // 16*9*64*64 = 589824
  const int b   = idx / (9 * COUT * CIN);
  const int r   = idx % (9 * COUT * CIN);
  const int tap = r / (COUT * CIN);
  const int r2  = r % (COUT * CIN);
  const int co  = r2 >> 6, ci = r2 & 63;
  const int kh  = tap / 3, kw = tap % 3;
  if (threadIdx.x < NEXP) g[threadIdx.x] = gates[b * NEXP + threadIdx.x];
  __syncthreads();
  float s = 0.f;
  for (int e = 0; e < NEXP; ++e)
    s += g[e] * W[(((e * COUT + co) * CIN + ci) * 3 + kh) * 3 + kw];
  weff[idx] = f2bf(s);
}

// ---------------- K5: WMMA conv + bias + store ----------------
// Grid: (H/2=32, B=16); block 256 = 8 waves. Per WG: output rows h0,h0+1 of
// batch b, all 64 Cout x 64 w = 32 WMMA tiles. Wave = (h_local, co-pair,
// w-pair) quad of 4 accumulators so each A/B fragment feeds 2 WMMAs.
__global__ __launch_bounds__(256) void k_conv(const float* __restrict__ x,
                                              const unsigned short* __restrict__ weff,
                                              const float* __restrict__ beff,
                                              float* __restrict__ y) {
  __shared__ __align__(16) unsigned short xs[XS_ELEMS];
  const int h0 = blockIdx.x * 2, b = blockIdx.y, tid = threadIdx.x;

  // ---- stage 4 input rows (h0-1..h0+2) transposed to [r][w1][ci] bf16 ----
  // Task = (ci-group of 8, r, w): 8 w-coalesced b32 loads -> 1 ds_store_b128.
  for (int idx = tid; idx < 8 * 4 * WW; idx += 256) {
    const int w = idx & 63;
    const int t2 = idx >> 6;
    const int r = t2 & 3;
    const int cg = t2 >> 2;                 // ci group: 8 channels
    const int hin = h0 - 1 + r;
    union { unsigned short h[8]; uint4 q; } pk;
    if (hin >= 0 && hin < HH) {
      const float* xp = x + (((long)b * CIN + cg * 8) * HH + hin) * WW + w;
#pragma unroll
      for (int j = 0; j < 8; ++j) pk.h[j] = f2bf(xp[j * (HH * WW)]);
    } else {
      pk.q = make_uint4(0u, 0u, 0u, 0u);
    }
    *(uint4*)&xs[(r * 66 + (w + 1)) * CI_STRIDE + cg * 8] = pk.q;
  }
  // zero the w-halo columns (w1 = 0 and 65), all 4 rows, b128 stores
  for (int idx = tid; idx < 4 * 2 * 8; idx += 256) {
    const int cg = idx & 7, t2 = idx >> 3;
    const int r = t2 >> 1, w1 = (t2 & 1) ? 65 : 0;
    *(uint4*)&xs[(r * 66 + w1) * CI_STRIDE + cg * 8] = make_uint4(0u, 0u, 0u, 0u);
  }
  __syncthreads();

  const int wv = tid >> 5, lane = tid & 31, lo = lane & 15, hi = lane >> 4;
  const int hl  = wv >> 2;                  // local output row 0/1
  const int co0 = ((wv >> 1) & 1) * 32;     // co tiles: co0, co0+16
  const int w0  = (wv & 1) * 32;            // w tiles:  w0,  w0+16
  const unsigned short* wbp = weff + b * (9 * COUT * CIN);

  v8f c00 = {0.f,0.f,0.f,0.f,0.f,0.f,0.f,0.f};
  v8f c01 = c00, c10 = c00, c11 = c00;

#pragma unroll
  for (int kh = 0; kh < 3; ++kh)
#pragma unroll
    for (int kw = 0; kw < 3; ++kw)
#pragma unroll
      for (int ck = 0; ck < 2; ++ck) {
        const int cib = ck * 32 + hi * 8;
        const int tap = kh * 3 + kw;
        // A fragments (weights): m = lo, k per 16-bit A ISA layout
        v16bf a0 = mk16(wbp + (tap * COUT + co0 + lo) * CIN + cib);
        v16bf a1 = mk16(wbp + (tap * COUT + co0 + 16 + lo) * CIN + cib);
        // B fragments (pixels): n = lo; input col = w + kw - 1 (+1 halo);
        // input row for output h0+hl with tap kh = slab row hl+kh (0..3)
        const int row = hl + kh;
        const int wc = w0 + lo + kw;
        v16bf b0 = mk16(&xs[(row * 66 + wc) * CI_STRIDE + cib]);
        v16bf b1 = mk16(&xs[(row * 66 + wc + 16) * CI_STRIDE + cib]);
        c00 = __builtin_amdgcn_wmma_f32_16x16x32_bf16(false, a0, false, b0, (short)0, c00, false, false);
        c01 = __builtin_amdgcn_wmma_f32_16x16x32_bf16(false, a0, false, b1, (short)0, c01, false, false);
        c10 = __builtin_amdgcn_wmma_f32_16x16x32_bf16(false, a1, false, b0, (short)0, c10, false, false);
        c11 = __builtin_amdgcn_wmma_f32_16x16x32_bf16(false, a1, false, b1, (short)0, c11, false, false);
      }

  // C/D layout: VGPR r: lanes 0-15 -> M=r, lanes 16-31 -> M=8+r; N = lane%16.
  const int h = h0 + hl;
#pragma unroll
  for (int r2 = 0; r2 < 8; ++r2) {
    const int coA = co0 + hi * 8 + r2;
    const int coB = coA + 16;
    const float bvA = beff[b * COUT + coA];
    const float bvB = beff[b * COUT + coB];
    float* ypA = y + (((long)b * COUT + coA) * HH + h) * WW;
    float* ypB = y + (((long)b * COUT + coB) * HH + h) * WW;
    ypA[w0 + lo]      = c00[r2] + bvA;
    ypA[w0 + 16 + lo] = c01[r2] + bvA;
    ypB[w0 + lo]      = c10[r2] + bvB;
    ypB[w0 + 16 + lo] = c11[r2] + bvB;
  }
}

// ---------------- host ----------------
extern "C" void kernel_launch(void* const* d_in, const int* in_sizes, int n_in,
                              void* d_out, int out_size, void* d_ws, size_t ws_size,
                              hipStream_t stream) {
  (void)in_sizes; (void)n_in; (void)ws_size;
  const float* x      = (const float*)d_in[0];   // (16,64,64,64)
  const float* w_gate = (const float*)d_in[1];   // (64,16)
  // d_in[2] = w_noise (unused, train=0)
  const float* W      = (const float*)d_in[3];   // (16,64,64,3,3)
  const float* bias   = (const float*)d_in[4];   // (16,64)
  float* y = (float*)d_out;
  float* loss_out = (float*)d_out + (out_size - 1);

  char* ws = (char*)d_ws;
  float* gate_x         = (float*)(ws + 0);        // 16*64 f32   = 4096 B
  float* gates          = (float*)(ws + 4096);     // 16*16 f32   = 1024 B
  float* beff           = (float*)(ws + 5120);     // 16*64 f32   = 4096 B
  unsigned short* weff  = (unsigned short*)(ws + 9216); // 589824 bf16 = 1179648 B

  k_gatex<<<BATCH * CIN, 256, 0, stream>>>(x, gate_x);
  k_gate<<<1, 256, 0, stream>>>(gate_x, w_gate, gates, loss_out);
  k_beff<<<(BATCH * COUT) / 256, 256, 0, stream>>>(bias, gates, beff);
  k_weff<<<(BATCH * 9 * COUT * CIN) / 256, 256, 0, stream>>>(W, gates, weff);
  k_conv<<<dim3(HH / 2, BATCH), 256, 0, stream>>>(x, weff, beff, y);
}
